// SwitchMLP_89189290868940
// MI455X (gfx1250) — matrix operand order repaired and checked by reading, not compiled, not run on hardware
//
#include <hip/hip_runtime.h>
#include <hip/hip_bf16.h>
#include <stdint.h>

#define N_TOK 2048
#define DDIM  1024
#define IDIM  2048
#define NEXP  8
#define TOPK  2
#define PAIRS (N_TOK * TOPK)
#define MTILE 32                       // row-tile (tokens) per workgroup
#define MAXROWS (PAIRS + NEXP * MTILE) // padded per-expert segments
#define MAXTILES (MAXROWS / MTILE)     // 136 row-tile slots

typedef __hip_bfloat16 bf16;
typedef __attribute__((ext_vector_type(16))) __bf16 v16bf;
typedef __attribute__((ext_vector_type(8)))  float  v8f;

union Frag { v16bf v; uint4 u[2]; };

__device__ __forceinline__ v8f wmma_bf16(const Frag& a, const Frag& b, v8f c) {
  return __builtin_amdgcn_wmma_f32_16x16x32_bf16(false, a.v, false, b.v,
                                                 (short)0, c, false, false);
}

// ---------------- fp32 -> bf16 conversion (weights live in 192MB L2 as bf16) --
__global__ __launch_bounds__(256) void k_cvt(const float* __restrict__ s,
                                             bf16* __restrict__ d, int n) {
  int stride = gridDim.x * blockDim.x;
  for (int i = blockIdx.x * blockDim.x + threadIdx.x; i * 4 < n; i += stride) {
    float4 v = *(const float4*)(s + (size_t)i * 4);
    union { bf16 h[4]; uint2 u; } o;
    o.h[0] = __float2bfloat16(v.x); o.h[1] = __float2bfloat16(v.y);
    o.h[2] = __float2bfloat16(v.z); o.h[3] = __float2bfloat16(v.w);
    *(uint2*)(d + (size_t)i * 4) = o.u;
  }
}

__global__ __launch_bounds__(256) void k_zero(float* __restrict__ p, int n) {
  int stride = gridDim.x * blockDim.x;
  for (int i = blockIdx.x * blockDim.x + threadIdx.x; i < n; i += stride)
    p[i] = 0.0f;
}

// ---------------- routing: build per-expert padded token lists ---------------
__global__ __launch_bounds__(256) void k_route(const int* __restrict__ idx64,
                                               const float* __restrict__ ew,
                                               int* __restrict__ toks,
                                               float* __restrict__ wl,
                                               int* __restrict__ offs) {
  __shared__ int cnt[NEXP], cur[NEXP], soff[NEXP + 1];
  const int t = threadIdx.x;
  if (t < NEXP) cnt[t] = 0;
  __syncthreads();
  for (int p = t; p < PAIRS; p += 256) {
    int e = idx64[2 * p] & (NEXP - 1);   // int64 indices, little-endian low dword
    atomicAdd(&cnt[e], 1);
  }
  __syncthreads();
  if (t == 0) {
    int o = 0;
    for (int e = 0; e < NEXP; ++e) {
      soff[e] = o; cur[e] = o;
      o += ((cnt[e] + MTILE - 1) / MTILE) * MTILE;  // pad to 32-row tiles
    }
    soff[NEXP] = o;
    for (int e = 0; e <= NEXP; ++e) offs[e] = soff[e];
  }
  __syncthreads();
  for (int s = t; s < soff[NEXP]; s += 256) { toks[s] = 0; wl[s] = 0.0f; }
  __threadfence_block();
  __syncthreads();
  for (int p = t; p < PAIRS; p += 256) {
    int e = idx64[2 * p] & (NEXP - 1);
    int pos = atomicAdd(&cur[e], 1);
    toks[pos] = p >> 1;                  // token id
    wl[pos]   = ew[p];                   // routing weight
  }
}

// ---------------- gate/up GEMM + SiLU fuse (bf16 WMMA, double-buffered) ------
__global__ __launch_bounds__(256) void k_gateup(const bf16* __restrict__ xb,
                                                const bf16* __restrict__ wgb,
                                                const bf16* __restrict__ wub,
                                                const int*  __restrict__ toks,
                                                const int*  __restrict__ offs,
                                                bf16* __restrict__ h) {
  __shared__ int s_off[NEXP + 1];
  __shared__ int s_tok[MTILE];
  __shared__ alignas(16) bf16 sA[2][MTILE * 48];
  __shared__ alignas(16) bf16 sBg[2][128 * 48];
  __shared__ alignas(16) bf16 sBu[2][128 * 48];

  const int t = threadIdx.x;
  if (t <= NEXP) s_off[t] = offs[t];
  __syncthreads();
  const int rowBase = blockIdx.x * MTILE;
  if (rowBase >= s_off[NEXP]) return;            // uniform early exit
  int e = 0;
  while (rowBase >= s_off[e + 1]) ++e;
  if (t < MTILE) s_tok[t] = toks[rowBase + t];
  __syncthreads();

  const int colBase = blockIdx.y * 128;
  const bf16* __restrict__ Wg = wgb + (size_t)e * DDIM * IDIM;
  const bf16* __restrict__ Wu = wub + (size_t)e * DDIM * IDIM;

  // staging assignments: A = 32 rows x 32 k (one uint2/thread);
  // B = 32 k-rows x 128 cols (two uint4/thread per matrix)
  const int ar  = t >> 3;
  const int ac  = (t & 7) << 2;
  const int br0 = t >> 4;          // 0..15
  const int br1 = br0 + 16;        // 16..31
  const int bc  = (t & 15) << 3;
  const size_t aoff = (size_t)s_tok[ar] * DDIM + ac;

  uint2 ra;
  uint4 rg0, rg1, ru0, ru1;
  ra  = *(const uint2*)(xb + aoff);
  rg0 = *(const uint4*)(Wg + (size_t)br0 * IDIM + colBase + bc);
  ru0 = *(const uint4*)(Wu + (size_t)br0 * IDIM + colBase + bc);
  rg1 = *(const uint4*)(Wg + (size_t)br1 * IDIM + colBase + bc);
  ru1 = *(const uint4*)(Wu + (size_t)br1 * IDIM + colBase + bc);

  const int wave = t >> 5, lane = t & 31;
  const int m16 = lane & 15, half = lane >> 4;
  const int cB = (wave << 4) + m16;

  v8f accg0 = {}, accg1 = {}, accu0 = {}, accu1 = {};

  int cur = 0;
  for (int kk = 0; kk < DDIM; kk += 32) {
    // commit staged registers into LDS buffer `cur`
    *(uint2*)&sA[cur][ar * 48 + ac] = ra;
    {
      const bf16* p0 = (const bf16*)&rg0;
      const bf16* p1 = (const bf16*)&rg1;
      const bf16* q0 = (const bf16*)&ru0;
      const bf16* q1 = (const bf16*)&ru1;
      #pragma unroll
      for (int j = 0; j < 8; ++j) {
        sBg[cur][(bc + j) * 48 + br0] = p0[j];
        sBg[cur][(bc + j) * 48 + br1] = p1[j];
        sBu[cur][(bc + j) * 48 + br0] = q0[j];
        sBu[cur][(bc + j) * 48 + br1] = q1[j];
      }
    }
    __syncthreads();
    // prefetch next K-slab into registers while WMMAs run below
    const int kn = kk + 32;
    if (kn < DDIM) {
      ra  = *(const uint2*)(xb + aoff + kn);
      rg0 = *(const uint4*)(Wg + (size_t)(kn + br0) * IDIM + colBase + bc);
      ru0 = *(const uint4*)(Wu + (size_t)(kn + br0) * IDIM + colBase + bc);
      rg1 = *(const uint4*)(Wg + (size_t)(kn + br1) * IDIM + colBase + bc);
      ru1 = *(const uint4*)(Wu + (size_t)(kn + br1) * IDIM + colBase + bc);
      if (kn + 32 < DDIM) {   // speculative L2 prefetch two slabs ahead
        __builtin_prefetch(Wg + (size_t)(kn + 32 + br0) * IDIM + colBase + bc, 0, 0);
        __builtin_prefetch(Wu + (size_t)(kn + 32 + br0) * IDIM + colBase + bc, 0, 0);
      }
    }
    Frag a0, a1, bg, bu;
    a0.u[0] = *(const uint4*)&sA[cur][m16 * 48 + half * 8];
    a0.u[1] = *(const uint4*)&sA[cur][m16 * 48 + 16 + half * 8];
    a1.u[0] = *(const uint4*)&sA[cur][(m16 + 16) * 48 + half * 8];
    a1.u[1] = *(const uint4*)&sA[cur][(m16 + 16) * 48 + 16 + half * 8];
    bg.u[0] = *(const uint4*)&sBg[cur][cB * 48 + half * 8];
    bg.u[1] = *(const uint4*)&sBg[cur][cB * 48 + 16 + half * 8];
    bu.u[0] = *(const uint4*)&sBu[cur][cB * 48 + half * 8];
    bu.u[1] = *(const uint4*)&sBu[cur][cB * 48 + 16 + half * 8];
    accg0 = wmma_bf16(a0, bg, accg0);
    accg1 = wmma_bf16(a1, bg, accg1);
    accu0 = wmma_bf16(a0, bu, accu0);
    accu1 = wmma_bf16(a1, bu, accu1);
    cur ^= 1;
  }

  const int col = colBase + cB;
  #pragma unroll
  for (int r = 0; r < 8; ++r) {
    const int m = r + (half << 3);
    {
      float g = accg0[r], u = accu0[r];
      float hv = g * u * __builtin_amdgcn_rcpf(1.0f + __expf(-g));
      h[(size_t)(rowBase + m) * IDIM + col] = __float2bfloat16(hv);
    }
    {
      float g = accg1[r], u = accu1[r];
      float hv = g * u * __builtin_amdgcn_rcpf(1.0f + __expf(-g));
      h[(size_t)(rowBase + 16 + m) * IDIM + col] = __float2bfloat16(hv);
    }
  }
}

// ---------------- down GEMM + weighted atomic scatter (double-buffered) ------
__global__ __launch_bounds__(256) void k_down(const bf16* __restrict__ h,
                                              const bf16* __restrict__ wdb,
                                              const int*  __restrict__ toks,
                                              const float* __restrict__ wl,
                                              const int*  __restrict__ offs,
                                              float* __restrict__ out) {
  __shared__ int s_off[NEXP + 1];
  __shared__ int s_tok[MTILE];
  __shared__ float s_w[MTILE];
  __shared__ alignas(16) bf16 sA[2][MTILE * 48];
  __shared__ alignas(16) bf16 sB[2][128 * 48];

  const int t = threadIdx.x;
  if (t <= NEXP) s_off[t] = offs[t];
  __syncthreads();
  const int rowBase = blockIdx.x * MTILE;
  if (rowBase >= s_off[NEXP]) return;
  int e = 0;
  while (rowBase >= s_off[e + 1]) ++e;
  if (t < MTILE) { s_tok[t] = toks[rowBase + t]; s_w[t] = wl[rowBase + t]; }
  __syncthreads();

  const int colBase = blockIdx.y * 128;
  const bf16* __restrict__ Wd = wdb + (size_t)e * IDIM * DDIM;

  const int ar  = t >> 3;
  const int ac  = (t & 7) << 2;
  const int br0 = t >> 4;
  const int br1 = br0 + 16;
  const int bc  = (t & 15) << 3;
  const size_t aoff = (size_t)(rowBase + ar) * IDIM + ac;

  uint2 ra;
  uint4 rb0, rb1;
  ra  = *(const uint2*)(h + aoff);
  rb0 = *(const uint4*)(Wd + (size_t)br0 * DDIM + colBase + bc);
  rb1 = *(const uint4*)(Wd + (size_t)br1 * DDIM + colBase + bc);

  const int wave = t >> 5, lane = t & 31;
  const int m16 = lane & 15, half = lane >> 4;
  const int cB = (wave << 4) + m16;

  v8f acc0 = {}, acc1 = {};

  int cur = 0;
  for (int kk = 0; kk < IDIM; kk += 32) {
    *(uint2*)&sA[cur][ar * 48 + ac] = ra;
    {
      const bf16* p0 = (const bf16*)&rb0;
      const bf16* p1 = (const bf16*)&rb1;
      #pragma unroll
      for (int j = 0; j < 8; ++j) {
        sB[cur][(bc + j) * 48 + br0] = p0[j];
        sB[cur][(bc + j) * 48 + br1] = p1[j];
      }
    }
    __syncthreads();
    const int kn = kk + 32;
    if (kn < IDIM) {
      ra  = *(const uint2*)(h + aoff + kn);
      rb0 = *(const uint4*)(Wd + (size_t)(kn + br0) * DDIM + colBase + bc);
      rb1 = *(const uint4*)(Wd + (size_t)(kn + br1) * DDIM + colBase + bc);
      if (kn + 32 < IDIM)
        __builtin_prefetch(Wd + (size_t)(kn + 32 + br0) * DDIM + colBase + bc, 0, 0);
    }
    Frag a0, a1, b;
    a0.u[0] = *(const uint4*)&sA[cur][m16 * 48 + half * 8];
    a0.u[1] = *(const uint4*)&sA[cur][m16 * 48 + 16 + half * 8];
    a1.u[0] = *(const uint4*)&sA[cur][(m16 + 16) * 48 + half * 8];
    a1.u[1] = *(const uint4*)&sA[cur][(m16 + 16) * 48 + 16 + half * 8];
    b.u[0] = *(const uint4*)&sB[cur][cB * 48 + half * 8];
    b.u[1] = *(const uint4*)&sB[cur][cB * 48 + 16 + half * 8];
    acc0 = wmma_bf16(a0, b, acc0);
    acc1 = wmma_bf16(a1, b, acc1);
    cur ^= 1;
  }

  const int col = colBase + cB;
  #pragma unroll
  for (int r = 0; r < 8; ++r) {
    const int m = r + (half << 3);
    atomicAdd(&out[(size_t)s_tok[m] * DDIM + col], acc0[r] * s_w[m]);
    atomicAdd(&out[(size_t)s_tok[m + 16] * DDIM + col], acc1[r] * s_w[m + 16]);
  }
}

// ---------------- launcher ---------------------------------------------------
extern "C" void kernel_launch(void* const* d_in, const int* in_sizes, int n_in,
                              void* d_out, int out_size, void* d_ws, size_t ws_size,
                              hipStream_t stream) {
  (void)in_sizes; (void)n_in; (void)out_size; (void)ws_size;
  const float* x   = (const float*)d_in[0];
  const float* ew  = (const float*)d_in[1];
  const float* wg  = (const float*)d_in[2];
  const float* wu  = (const float*)d_in[3];
  const float* wd  = (const float*)d_in[4];
  const int*   idx = (const int*)d_in[5];     // int64 expert indices
  float* out = (float*)d_out;

  char* ws = (char*)d_ws;
  size_t o = 0;
  bf16* xb   = (bf16*)(ws + o); o += (size_t)N_TOK * DDIM * 2;
  bf16* wgb  = (bf16*)(ws + o); o += (size_t)NEXP * DDIM * IDIM * 2;
  bf16* wub  = (bf16*)(ws + o); o += (size_t)NEXP * DDIM * IDIM * 2;
  bf16* wdb  = (bf16*)(ws + o); o += (size_t)NEXP * IDIM * DDIM * 2;
  bf16* hbuf = (bf16*)(ws + o); o += (size_t)MAXROWS * IDIM * 2;
  int*  toks = (int*)(ws + o);  o += (size_t)MAXROWS * 4;
  float* wl  = (float*)(ws + o); o += (size_t)MAXROWS * 4;
  int*  offs = (int*)(ws + o);  o += 64;

  k_cvt<<<2048, 256, 0, stream>>>(x,  xb,  N_TOK * DDIM);
  k_cvt<<<4096, 256, 0, stream>>>(wg, wgb, NEXP * DDIM * IDIM);
  k_cvt<<<4096, 256, 0, stream>>>(wu, wub, NEXP * DDIM * IDIM);
  k_cvt<<<4096, 256, 0, stream>>>(wd, wdb, NEXP * IDIM * DDIM);
  k_route<<<1, 256, 0, stream>>>(idx, ew, toks, wl, offs);
  k_zero<<<2048, 256, 0, stream>>>(out, N_TOK * DDIM);

  dim3 g2(MAXTILES, IDIM / 128);
  k_gateup<<<g2, 256, 0, stream>>>(xb, wgb, wub, toks, offs, hbuf);
  dim3 g3(MAXTILES, DDIM / 128);
  k_down<<<g3, 256, 0, stream>>>(hbuf, wdb, toks, wl, offs, out);
}
